// LaBraM_74509092651587
// MI455X (gfx1250) — compile-verified, matrix operand order checked by reading, not run on hardware
//
#include <hip/hip_runtime.h>
#include <hip/hip_bf16.h>
#include <math.h>

typedef __bf16 bf16;
typedef __attribute__((ext_vector_type(16))) __bf16       v16bf;
typedef __attribute__((ext_vector_type(8)))  float        v8f;
typedef __attribute__((ext_vector_type(4)))  unsigned int v4u;

#define DEV __device__ __forceinline__

static constexpr int BATCH = 8;
static constexpr int NTOK  = 513;
static constexpr int BN    = BATCH * NTOK;   // 4104 token rows
static constexpr int BNP   = 4128;           // padded to 32 (M supertile)
static constexpr int DMODEL= 200;
static constexpr int DP    = 224;            // padded embed dim (7 x 32)
static constexpr int NH    = 10;
static constexpr int HD    = 20;
static constexpr int NKP   = 544;            // padded sequence (17 x 32)
static constexpr int QKV_LD= 608;            // padded 3*D
static constexpr int FF    = 800;            // MLP hidden (25 x 32)
static constexpr int NRELR = 1908;
// B-operand row paddings (N-tile-group sweep = grid.y * 64 rows, zero-filled)
static constexpr int QKV_RP = 640;   // grid.y = 10
static constexpr int PRJ_RP = 256;   // grid.y = 4
static constexpr int FC1_RP = 832;   // grid.y = 13
static constexpr int FC2_RP = 256;   // grid.y = 4

// ---------------------------------------------------------------- fragments
// A-operand (M=16 x K=32 bf16): lane L -> row (L&15); lane holds K offsets
// {h*8..h*8+7} and {16+h*8..+7}, h = L>>4 -> two contiguous 16B chunks.
// B-operand supplied as N x K (natural [out][in] weights), mirrored layout.
DEV v16bf load_frag(const bf16* p, int ld) {
  int lane = threadIdx.x & 31;
  const bf16* q = p + (size_t)(lane & 15) * ld + ((lane >> 4) << 3);
  union { v16bf f; v4u u[2]; } t;
  t.u[0] = *reinterpret_cast<const v4u*>(q);
  t.u[1] = *reinterpret_cast<const v4u*>(q + 16);
  return t.f;
}

DEV v8f wmma_bf16(v16bf a, v16bf b, v8f c) {
  return __builtin_amdgcn_wmma_f32_16x16x32_bf16(false, a, false, b,
                                                 (short)0, c, false, false);
}

DEV float gelu_f(float x) { return 0.5f * x * (1.0f + erff(x * 0.70710678118654752f)); }

DEV float wave_sum(float v) {
  for (int o = 16; o; o >>= 1) v += __shfl_xor(v, o);
  return v;
}

// ---------------------------------------------------------------- weight prep
__global__ void pad_w_k(const float* __restrict__ W, bf16* __restrict__ Wb,
                        int layers, int rows, int cols, int rpad, int cpad) {
  long i = (long)blockIdx.x * blockDim.x + threadIdx.x;
  long total = (long)layers * rpad * cpad;
  if (i >= total) return;
  int c = (int)(i % cpad);
  int r = (int)((i / cpad) % rpad);
  int l = (int)(i / ((long)cpad * rpad));
  float v = (r < rows && c < cols) ? W[((size_t)l * rows + r) * cols + c] : 0.0f;
  Wb[i] = (bf16)v;
}

// ---------------------------------------------------------------- conv front
__global__ void conv1_k(const float* __restrict__ x, const float* __restrict__ w,
                        const float* __restrict__ b, float* __restrict__ out) {
  int i = blockIdx.x * blockDim.x + threadIdx.x;         // (B,8,512,25)
  if (i >= BATCH * 8 * 512 * 25) return;
  int wo = i % 25, row = (i / 25) % 512, c = (i / 12800) % 8, bb = i / 102400;
  float s = b[c];
  int wi0 = wo * 8 - 7;
  const float* xp = x + ((size_t)bb * 512 + row) * 200;
  const float* wp = w + c * 15;
#pragma unroll
  for (int t = 0; t < 15; ++t) {
    int wi = wi0 + t;
    if (wi >= 0 && wi < 200) s += xp[wi] * wp[t];
  }
  out[i] = s;
}

__global__ void conv3x1_k(const float* __restrict__ in, const float* __restrict__ w,
                          const float* __restrict__ b, float* __restrict__ out) {
  int i = blockIdx.x * blockDim.x + threadIdx.x;         // (B,8,512,25)
  if (i >= BATCH * 8 * 512 * 25) return;
  int wo = i % 25, row = (i / 25) % 512, c = (i / 12800) % 8, bb = i / 102400;
  float s = b[c];
  for (int ci = 0; ci < 8; ++ci) {
    const float* ip = in + ((size_t)(bb * 8 + ci) * 512 + row) * 25;
    const float* wp = w + (c * 8 + ci) * 3;
#pragma unroll
    for (int t = 0; t < 3; ++t) {
      int wi = wo - 1 + t;
      if (wi >= 0 && wi < 25) s += ip[wi] * wp[t];
    }
  }
  out[i] = s;
}

__global__ void gn_stats_k(const float* __restrict__ X, float* __restrict__ stats) {
  // blocks = B*4 ; group = 2 consecutive channels = 25600 contiguous floats
  int bg = blockIdx.x;
  const float* base = X + ((size_t)(bg >> 2) * 8 + (bg & 3) * 2) * 12800;
  float s = 0.f, s2 = 0.f;
  for (int i = threadIdx.x; i < 25600; i += blockDim.x) {
    float v = base[i]; s += v; s2 += v * v;
  }
  __shared__ float rs[256], rs2[256];
  rs[threadIdx.x] = s; rs2[threadIdx.x] = s2;
  __syncthreads();
  for (int o = 128; o; o >>= 1) {
    if (threadIdx.x < o) { rs[threadIdx.x] += rs[threadIdx.x + o]; rs2[threadIdx.x] += rs2[threadIdx.x + o]; }
    __syncthreads();
  }
  if (threadIdx.x == 0) {
    float m = rs[0] / 25600.0f;
    stats[bg * 2] = m;
    stats[bg * 2 + 1] = rs2[0] / 25600.0f - m * m;
  }
}

__global__ void gn_apply_k(float* __restrict__ X, const float* __restrict__ stats,
                           const float* __restrict__ gw, const float* __restrict__ gb) {
  int i = blockIdx.x * blockDim.x + threadIdx.x;
  if (i >= BATCH * 8 * 512 * 25) return;
  int c = (i / 12800) % 8, bb = i / 102400;
  int g = c >> 1;
  float m = stats[(bb * 4 + g) * 2], var = stats[(bb * 4 + g) * 2 + 1];
  float v = (X[i] - m) * rsqrtf(var + 1e-5f) * gw[c] + gb[c];
  X[i] = gelu_f(v);
}

__global__ void embed_k(const float* __restrict__ conv, const float* __restrict__ pos,
                        const float* __restrict__ te, float* __restrict__ h) {
  int i = blockIdx.x * blockDim.x + threadIdx.x;         // B*512*200
  if (i >= BATCH * 512 * 200) return;
  int d = i % 200, p = (i / 200) % 512, bb = i / 102400;
  int c = d & 7, t = d >> 3;
  float v = conv[((size_t)(bb * 8 + c) * 512 + p) * 25 + t];
  v += pos[(1 + (p >> 3)) * 200 + d] + te[(p & 7) * 200 + d];
  h[((size_t)bb * NTOK + 1 + p) * DP + d] = v;
}

__global__ void cls_k(const float* __restrict__ cls, const float* __restrict__ pos,
                      float* __restrict__ h) {
  int i = blockIdx.x * blockDim.x + threadIdx.x;         // B*200
  if (i >= BATCH * 200) return;
  int d = i % 200, bb = i / 200;
  h[(size_t)bb * NTOK * DP + d] = cls[d] + pos[d];
}

// ---------------------------------------------------------------- layernorm
__global__ void ln_rows_k(const float* __restrict__ X, const float* __restrict__ g,
                          const float* __restrict__ bta, bf16* __restrict__ Y) {
  int row = blockIdx.x * blockDim.x + threadIdx.x;
  if (row >= BN) return;
  const float* x = X + (size_t)row * DP;
  float s = 0.f;
  for (int d = 0; d < DMODEL; ++d) s += x[d];
  float m = s / DMODEL;
  float v = 0.f;
  for (int d = 0; d < DMODEL; ++d) { float dv = x[d] - m; v += dv * dv; }
  float rs = rsqrtf(v / DMODEL + 1e-6f);
  bf16* y = Y + (size_t)row * DP;
  for (int d = 0; d < DMODEL; ++d) y[d] = (bf16)((x[d] - m) * rs * g[d] + bta[d]);
}

// ---------------------------------------------------------------- WMMA GEMM
// 32x64 output tile per wave: 2 A-frags x 4 B-frags -> 8 WMMAs per K-step.
// C[m][n] = sum_k A[m][k] * B[n][k]   (B supplied N x K, rows zero-padded to
// the full grid.y*64 sweep so B-frag loads are unguarded).
// mode 0: raw fp32 store            (QKV)
// mode 1: bf16 store of gelu(+bias) (fc1) — pad rows stored as 0
// mode 2: fp32 residual: out += gamma[n]*(acc+bias[n])   (proj / fc2)
__global__ void gemm_wmma_k(const bf16* __restrict__ A, int lda,
                            const bf16* __restrict__ B, int ldb, int ksteps,
                            const float* __restrict__ bias,
                            const float* __restrict__ gamma,
                            float* __restrict__ outF, bf16* __restrict__ outB,
                            int ldc, int Mvalid, int Nvalid, int mode) {
  int mt = blockIdx.x, ntg = blockIdx.y;
  const bf16* Ab = A + (size_t)mt * 32 * lda;
  const bf16* Bb = B + (size_t)ntg * 64 * ldb;
  v8f acc[2][4];
#pragma unroll
  for (int mi = 0; mi < 2; ++mi)
#pragma unroll
    for (int ni = 0; ni < 4; ++ni)
      acc[mi][ni] = (v8f){0.f, 0.f, 0.f, 0.f, 0.f, 0.f, 0.f, 0.f};

  for (int ks = 0; ks < ksteps; ++ks) {
    v16bf a0 = load_frag(Ab + ks * 32, lda);
    v16bf a1 = load_frag(Ab + (size_t)16 * lda + ks * 32, lda);
    v16bf b0 = load_frag(Bb + ks * 32, ldb);
    v16bf b1 = load_frag(Bb + (size_t)16 * ldb + ks * 32, ldb);
    v16bf b2 = load_frag(Bb + (size_t)32 * ldb + ks * 32, ldb);
    v16bf b3 = load_frag(Bb + (size_t)48 * ldb + ks * 32, ldb);
    acc[0][0] = wmma_bf16(a0, b0, acc[0][0]);
    acc[0][1] = wmma_bf16(a0, b1, acc[0][1]);
    acc[0][2] = wmma_bf16(a0, b2, acc[0][2]);
    acc[0][3] = wmma_bf16(a0, b3, acc[0][3]);
    acc[1][0] = wmma_bf16(a1, b0, acc[1][0]);
    acc[1][1] = wmma_bf16(a1, b1, acc[1][1]);
    acc[1][2] = wmma_bf16(a1, b2, acc[1][2]);
    acc[1][3] = wmma_bf16(a1, b3, acc[1][3]);
  }

  int lane = threadIdx.x & 31;
  int nl = lane & 15, hf = lane >> 4;
#pragma unroll
  for (int mi = 0; mi < 2; ++mi) {
#pragma unroll
    for (int j = 0; j < 8; ++j) {
      int m = mt * 32 + mi * 16 + j + 8 * hf;
#pragma unroll
      for (int ni = 0; ni < 4; ++ni) {
        int n = ntg * 64 + ni * 16 + nl;
        float v = acc[mi][ni][j];
        if (mode == 0) {
          if (m < Mvalid && n < Nvalid) outF[(size_t)m * ldc + n] = v;
        } else if (mode == 1) {
          if (n < ldc) {
            float r = 0.f;
            if (m < Mvalid && n < Nvalid) r = gelu_f(v + bias[n]);
            outB[(size_t)m * ldc + n] = (bf16)r;
          }
        } else {
          if (m < Mvalid && n < Nvalid)
            outF[(size_t)m * ldc + n] += gamma[n] * (v + bias[n]);
        }
      }
    }
  }
}

// ---------------------------------------------------------------- qkv reshape
__global__ void qkv_reshape_k(const float* __restrict__ qkvO, bf16* __restrict__ Qb,
                              bf16* __restrict__ Kb, bf16* __restrict__ Vt,
                              float scale) {
  int i = blockIdx.x * blockDim.x + threadIdx.x;     // BH * NKP * 32
  if (i >= BATCH * NH * NKP * 32) return;
  int dd = i & 31, nn = (i >> 5) % NKP, bh = i / (NKP * 32);
  int bb = bh / NH, hh = bh % NH;
  float qv = 0.f, kv = 0.f, vv = 0.f;
  if (nn < NTOK && dd < HD) {
    size_t row = (size_t)(bb * NTOK + nn) * QKV_LD;
    int col = hh * HD + dd;
    qv = qkvO[row + col] * scale;
    kv = qkvO[row + 200 + col];
    vv = qkvO[row + 400 + col];
  }
  Qb[i] = (bf16)qv;
  Kb[i] = (bf16)kv;
  Vt[(size_t)bh * 32 * NKP + (size_t)dd * NKP + nn] = (bf16)vv;
}

// ---------------------------------------------------------------- attention
// One wave per (batch*head, 16-query tile).  S via WMMA -> LDS, inline
// relative-position bias gather, softmax, P*V via WMMA, 1/sum at store.
__global__ void attn_k(const bf16* __restrict__ Qb, const bf16* __restrict__ Kb,
                       const bf16* __restrict__ Vt,
                       const float* __restrict__ bias_table,
                       const int* __restrict__ rel_idx,
                       bf16* __restrict__ attnOut, int layer) {
  int bh = blockIdx.x, qt = blockIdx.y;
  int bb = bh / NH, hh = bh % NH;
  int lane = threadIdx.x & 31;
  const bf16* Qbase = Qb + (size_t)bh * NKP * 32;
  const bf16* Kbase = Kb + (size_t)bh * NKP * 32;
  const bf16* Vbase = Vt + (size_t)bh * 32 * NKP;

  __shared__ float Sf[16 * NKP];
  __shared__ bf16  Pb[16 * NKP];
  __shared__ float ssum[16];

  v16bf qf = load_frag(Qbase + (size_t)qt * 16 * 32, 32);

  for (int kt = 0; kt < NKP / 16; ++kt) {
    v16bf kf = load_frag(Kbase + (size_t)kt * 16 * 32, 32);
    v8f acc = {0.f, 0.f, 0.f, 0.f, 0.f, 0.f, 0.f, 0.f};
    acc = wmma_bf16(qf, kf, acc);
    int n = lane & 15, hf = lane >> 4;
#pragma unroll
    for (int j = 0; j < 8; ++j) Sf[(j + 8 * hf) * NKP + kt * 16 + n] = acc[j];
  }
  __syncthreads();

  {
    int r = lane & 15, part = lane >> 4;
    int q = qt * 16 + r;
    int k0 = part * (NKP / 2), k1 = k0 + NKP / 2;
    const float* tbl = bias_table + (size_t)layer * NRELR * NH + hh;
    float mx = -3.0e38f;
    if (q < NTOK) {
      const int* irow = rel_idx + (size_t)q * NTOK;
      for (int k = k0; k < k1; ++k) {
        float s;
        if (k < NTOK) s = Sf[r * NKP + k] + tbl[irow[k] * NH];
        else          s = -3.0e38f;
        Sf[r * NKP + k] = s;
        mx = fmaxf(mx, s);
      }
    }
    mx = fmaxf(mx, __shfl_xor(mx, 16));
    float sum = 0.f;
    if (q < NTOK) {
      for (int k = k0; k < k1; ++k) {
        float e = __expf(Sf[r * NKP + k] - mx);
        Pb[r * NKP + k] = (bf16)e;
        sum += e;
      }
    } else {
      for (int k = k0; k < k1; ++k) Pb[r * NKP + k] = (bf16)0.0f;
    }
    sum += __shfl_xor(sum, 16);
    if (part == 0) ssum[r] = sum;
  }
  __syncthreads();

  v8f o0 = {0.f, 0.f, 0.f, 0.f, 0.f, 0.f, 0.f, 0.f};
  v8f o1 = {0.f, 0.f, 0.f, 0.f, 0.f, 0.f, 0.f, 0.f};
  for (int ks = 0; ks < NKP / 32; ++ks) {
    v16bf pf = load_frag(&Pb[0] + ks * 32, NKP);
    v16bf v0 = load_frag(Vbase + ks * 32, NKP);            // d rows 0..15
    v16bf v1 = load_frag(Vbase + 16 * NKP + ks * 32, NKP); // d rows 16..31
    o0 = wmma_bf16(pf, v0, o0);
    o1 = wmma_bf16(pf, v1, o1);
  }

  int n = lane & 15, hf = lane >> 4;
#pragma unroll
  for (int j = 0; j < 8; ++j) {
    int m = j + 8 * hf;
    int q = qt * 16 + m;
    if (q < NTOK) {
      float s = ssum[m];
      float inv = (s > 0.f) ? 1.0f / s : 0.f;
      size_t row = (size_t)(bb * NTOK + q) * DP + hh * HD;
      attnOut[row + n] = (bf16)(o0[j] * inv);
      if (n < HD - 16) attnOut[row + 16 + n] = (bf16)(o1[j] * inv);
    }
  }
}

// ---------------------------------------------------------------- pool + head
__global__ void pool_k(const float* __restrict__ h, float* __restrict__ pooled) {
  int i = blockIdx.x * blockDim.x + threadIdx.x;   // B*200
  if (i >= BATCH * 200) return;
  int bb = i / 200, d = i % 200;
  float s = 0.f;
  for (int t = 0; t < 512; ++t) s += h[((size_t)bb * NTOK + 1 + t) * DP + d];
  pooled[i] = s / 512.0f;
}

__global__ void head_k(const float* __restrict__ pooled, const float* __restrict__ g,
                       const float* __restrict__ bta, const float* __restrict__ hw,
                       const float* __restrict__ hb, float* __restrict__ out) {
  int bb = blockIdx.x;
  int lane = threadIdx.x & 31;
  const float* p = pooled + bb * 200;
  float s = 0.f;
  for (int d = lane; d < 200; d += 32) s += p[d];
  float m = wave_sum(s) / 200.0f;
  float v = 0.f;
  for (int d = lane; d < 200; d += 32) { float dv = p[d] - m; v += dv * dv; }
  float rs = rsqrtf(wave_sum(v) / 200.0f + 1e-6f);
  for (int j = 0; j < 4; ++j) {
    float a = 0.f;
    for (int d = lane; d < 200; d += 32)
      a += ((p[d] - m) * rs * g[d] + bta[d]) * hw[j * 200 + d];
    a = wave_sum(a);
    if (lane == 0) out[bb * 4 + j] = a + hb[j];
  }
}

// ---------------------------------------------------------------- launcher
extern "C" void kernel_launch(void* const* d_in, const int* in_sizes, int n_in,
                              void* d_out, int out_size, void* d_ws, size_t ws_size,
                              hipStream_t stream) {
  (void)in_sizes; (void)n_in; (void)out_size; (void)ws_size;
  const float* x        = (const float*)d_in[0];
  const float* conv1_w  = (const float*)d_in[1];
  const float* conv1_b  = (const float*)d_in[2];
  const float* gn1_g    = (const float*)d_in[3];
  const float* gn1_b    = (const float*)d_in[4];
  const float* conv2_w  = (const float*)d_in[5];
  const float* conv2_b  = (const float*)d_in[6];
  const float* gn2_g    = (const float*)d_in[7];
  const float* gn2_b    = (const float*)d_in[8];
  const float* conv3_w  = (const float*)d_in[9];
  const float* conv3_b  = (const float*)d_in[10];
  const float* gn3_g    = (const float*)d_in[11];
  const float* gn3_b    = (const float*)d_in[12];
  const float* cls_tok  = (const float*)d_in[13];
  const float* pos_emb  = (const float*)d_in[14];
  const float* time_emb = (const float*)d_in[15];
  const float* ln1_g    = (const float*)d_in[16];
  const float* ln1_b    = (const float*)d_in[17];
  const float* qkv_w    = (const float*)d_in[18];
  const float* rel_tab  = (const float*)d_in[19];
  const float* proj_w   = (const float*)d_in[20];
  const float* proj_b   = (const float*)d_in[21];
  const float* gamma1   = (const float*)d_in[22];
  const float* ln2_g    = (const float*)d_in[23];
  const float* ln2_b    = (const float*)d_in[24];
  const float* fc1_w    = (const float*)d_in[25];
  const float* fc1_b    = (const float*)d_in[26];
  const float* fc2_w    = (const float*)d_in[27];
  const float* fc2_b    = (const float*)d_in[28];
  const float* gamma2   = (const float*)d_in[29];
  const float* fcn_g    = (const float*)d_in[30];
  const float* fcn_b    = (const float*)d_in[31];
  const float* head_w   = (const float*)d_in[32];
  const float* head_b   = (const float*)d_in[33];
  const int*   rel_idx  = (const int*)d_in[34];

  char* w = (char*)d_ws;
  auto alloc = [&](size_t bytes) {
    char* p = w;
    w += (bytes + 255) & ~(size_t)255;
    return p;
  };
  bf16* qkvWb   = (bf16*)alloc((size_t)12 * QKV_RP * DP * 2);
  bf16* projWb  = (bf16*)alloc((size_t)12 * PRJ_RP * DP * 2);
  bf16* fc1Wb   = (bf16*)alloc((size_t)12 * FC1_RP * DP * 2);
  bf16* fc2Wb   = (bf16*)alloc((size_t)12 * FC2_RP * FF * 2);
  float* hbuf   = (float*)alloc((size_t)BNP * DP * 4);
  bf16* ybuf    = (bf16*)alloc((size_t)BNP * DP * 2);
  float* qkvO   = (float*)alloc((size_t)BNP * QKV_LD * 4);
  bf16* Qb      = (bf16*)alloc((size_t)BATCH * NH * NKP * 32 * 2);
  bf16* Kb      = (bf16*)alloc((size_t)BATCH * NH * NKP * 32 * 2);
  bf16* Vt      = (bf16*)alloc((size_t)BATCH * NH * 32 * NKP * 2);
  bf16* attnOut = (bf16*)alloc((size_t)BNP * DP * 2);
  bf16* mlpH    = (bf16*)alloc((size_t)BNP * FF * 2);
  float* stats  = (float*)alloc(32 * 2 * 4);
  float* pooled = (float*)alloc((size_t)BATCH * 200 * 4);
  // conv scratch aliases qkvO (conv pipeline finishes before QKV GEMM runs)
  float* c1 = qkvO;                                  // 8*8*512*25 f32 = 3.3MB
  float* c2 = (float*)((char*)qkvO + 5 * 1024 * 1024);

  auto cdiv = [](long a, long b) { return (unsigned)((a + b - 1) / b); };

  // ---- weight prep (bf16, padded) ----
  pad_w_k<<<cdiv((long)12 * QKV_RP * DP, 256), 256, 0, stream>>>(qkv_w, qkvWb, 12, 600, 200, QKV_RP, DP);
  pad_w_k<<<cdiv((long)12 * PRJ_RP * DP, 256), 256, 0, stream>>>(proj_w, projWb, 12, 200, 200, PRJ_RP, DP);
  pad_w_k<<<cdiv((long)12 * FC1_RP * DP, 256), 256, 0, stream>>>(fc1_w, fc1Wb, 12, 800, 200, FC1_RP, DP);
  pad_w_k<<<cdiv((long)12 * FC2_RP * FF, 256), 256, 0, stream>>>(fc2_w, fc2Wb, 12, 200, 800, FC2_RP, FF);

  hipMemsetAsync(hbuf, 0, (size_t)BNP * DP * 4, stream);
  hipMemsetAsync(ybuf, 0, (size_t)BNP * DP * 2, stream);
  hipMemsetAsync(attnOut, 0, (size_t)BNP * DP * 2, stream);

  // ---- temporal conv patch embed ----
  const int CE = BATCH * 8 * 512 * 25;
  conv1_k<<<cdiv(CE, 256), 256, 0, stream>>>(x, conv1_w, conv1_b, c1);
  gn_stats_k<<<32, 256, 0, stream>>>(c1, stats);
  gn_apply_k<<<cdiv(CE, 256), 256, 0, stream>>>(c1, stats, gn1_g, gn1_b);
  conv3x1_k<<<cdiv(CE, 256), 256, 0, stream>>>(c1, conv2_w, conv2_b, c2);
  gn_stats_k<<<32, 256, 0, stream>>>(c2, stats);
  gn_apply_k<<<cdiv(CE, 256), 256, 0, stream>>>(c2, stats, gn2_g, gn2_b);
  conv3x1_k<<<cdiv(CE, 256), 256, 0, stream>>>(c2, conv3_w, conv3_b, c1);
  gn_stats_k<<<32, 256, 0, stream>>>(c1, stats);
  gn_apply_k<<<cdiv(CE, 256), 256, 0, stream>>>(c1, stats, gn3_g, gn3_b);
  embed_k<<<cdiv(BATCH * 512 * 200, 256), 256, 0, stream>>>(c1, pos_emb, time_emb, hbuf);
  cls_k<<<cdiv(BATCH * 200, 256), 256, 0, stream>>>(cls_tok, pos_emb, hbuf);

  const float scale = 0.22360679774997896f;  // HD^-0.5
  const int MT32 = BNP / 32;                 // 129
  const int RSH = BATCH * NH * NKP * 32;

  for (int l = 0; l < 12; ++l) {
    ln_rows_k<<<cdiv(BN, 256), 256, 0, stream>>>(hbuf, ln1_g + l * 200, ln1_b + l * 200, ybuf);
    gemm_wmma_k<<<dim3(MT32, QKV_RP / 64), 32, 0, stream>>>(
        ybuf, DP, qkvWb + (size_t)l * QKV_RP * DP, DP, DP / 32,
        nullptr, nullptr, qkvO, nullptr, QKV_LD, BN, 600, 0);
    qkv_reshape_k<<<cdiv(RSH, 256), 256, 0, stream>>>(qkvO, Qb, Kb, Vt, scale);
    attn_k<<<dim3(BATCH * NH, 33), 32, 0, stream>>>(Qb, Kb, Vt, rel_tab, rel_idx, attnOut, l);
    gemm_wmma_k<<<dim3(MT32, PRJ_RP / 64), 32, 0, stream>>>(
        attnOut, DP, projWb + (size_t)l * PRJ_RP * DP, DP, DP / 32,
        proj_b + l * 200, gamma1 + l * 200, hbuf, nullptr, DP, BN, 200, 2);
    ln_rows_k<<<cdiv(BN, 256), 256, 0, stream>>>(hbuf, ln2_g + l * 200, ln2_b + l * 200, ybuf);
    gemm_wmma_k<<<dim3(MT32, FC1_RP / 64), 32, 0, stream>>>(
        ybuf, DP, fc1Wb + (size_t)l * FC1_RP * DP, DP, DP / 32,
        fc1_b + l * 800, nullptr, nullptr, mlpH, FF, BN, 800, 1);
    gemm_wmma_k<<<dim3(MT32, FC2_RP / 64), 32, 0, stream>>>(
        mlpH, FF, fc2Wb + (size_t)l * FC2_RP * FF, FF, FF / 32,
        fc2_b + l * 200, gamma2 + l * 200, hbuf, nullptr, DP, BN, 200, 2);
  }

  pool_k<<<cdiv(BATCH * 200, 256), 256, 0, stream>>>(hbuf, pooled);
  head_k<<<BATCH, 32, 0, stream>>>(pooled, fcn_g, fcn_b, head_w, head_b, (float*)d_out);
}
